// RegionLayer_33062658245169
// MI455X (gfx1250) — matrix operand order repaired and checked by reading, not compiled, not run on hardware
//
#include <hip/hip_runtime.h>

// ---- CDNA5 WMMA types --------------------------------------------------
typedef __attribute__((ext_vector_type(16))) __bf16        v16bf;
typedef __attribute__((ext_vector_type(8)))  float         v8f;
typedef __attribute__((ext_vector_type(4)))  unsigned int  u32x4;

union Frag {              // 8 VGPRs viewed as two 16B chunks or one bf16x16
    u32x4 q[2];
    v16bf v;
};

// ---- problem constants -------------------------------------------------
#define B_   16
#define C_   256
#define H_   64
#define W_   64
#define PH   8
#define PW   8
#define EPS_ 1e-5f

__device__ __forceinline__ unsigned short f2bf(float f) {
    unsigned int u = __builtin_bit_cast(unsigned int, f);
    u += 0x7FFFu + ((u >> 16) & 1u);          // round-to-nearest-even
    return (unsigned short)(u >> 16);
}

// Block tile: M=128 (2 batches x 64 px), N=128 (co half), K loop = 8 x 32ci x 9 taps.
// 8 waves = 2(M) x 4(N); wave tile 64x32 -> acc[4][2] WMMA fragments.
__global__ __launch_bounds__(256)
void RegionLayer_33062658245169_kernel(const float* __restrict__ x,
                                       const float* __restrict__ gamma,
                                       const float* __restrict__ beta,
                                       const float* __restrict__ mean,
                                       const float* __restrict__ var,
                                       const float* __restrict__ convw,
                                       const float* __restrict__ convb,
                                       float* __restrict__ out)
{
    __shared__ float s_scale[C_];
    __shared__ float s_shift[C_];
    __shared__ float s_cb[C_];
    // padded activation patch: [b_local(2)][yy(10)][xx(10)][ci(32)] bf16
    __shared__ __align__(16) unsigned short sA[2 * 10 * 10 * 32];     // 12.5 KB
    // weights, all 9 taps: [co_local(128)][tap(9)][ci(32)] bf16
    __shared__ __align__(16) unsigned short sB[128 * 9 * 32];         // 72 KB

    const int tid = threadIdx.x;
    const int rg  = blockIdx.x >> 4;          // region 0..63
    const int wgM = (blockIdx.x >> 1) & 7;    // M block (2 batches) 0..7
    const int wgN = blockIdx.x & 1;           // co half 0..1
    const int gi  = rg >> 3, gj = rg & 7;
    const int net = gj * (gi + 1);            // networks[i*j + j]
    const int gy0 = gi * PH, gx0 = gj * PW;
    const int b0  = wgM * 2;                  // first batch of this M block
    const int co0 = wgN * 128;                // first out-channel of this WG

    // Per-channel BN scale/shift + conv bias for this net (256 threads, one pass)
    {
        const int c = tid;
        const float sc = gamma[net * C_ + c] * rsqrtf(var[net * C_ + c] + EPS_);
        s_scale[c] = sc;
        s_shift[c] = beta[net * C_ + c] - mean[net * C_ + c] * sc;
        s_cb[c]    = convb[net * C_ + c];
    }
    __syncthreads();

    const int lane  = tid & 31;
    const int wave  = tid >> 5;          // 0..7
    const int waveM = wave >> 2;         // 0..1 -> 64-row slab
    const int waveN = wave & 3;          // 0..3 -> 32-col slab
    const int half  = lane >> 4;         // 0..1
    const int l16   = lane & 15;

    v8f acc[4][2];
    const v8f vzero = {0.f, 0.f, 0.f, 0.f, 0.f, 0.f, 0.f, 0.f};
#pragma unroll
    for (int i = 0; i < 4; ++i)
#pragma unroll
        for (int j = 0; j < 2; ++j) acc[i][j] = vzero;

    // Per-lane LDS fragment base pointers (tap offsets are immediates).
    const unsigned short* app[4];
#pragma unroll
    for (int mt = 0; mt < 4; ++mt) {
        const int m  = waveM * 64 + mt * 16 + l16;   // local M row 0..127
        const int bl = m >> 6;
        const int y  = (m >> 3) & 7;
        const int xx = m & 7;
        app[mt] = sA + (((bl * 10 + y) * 10 + xx) * 32) + half * 8;
    }
    const unsigned short* bpp[2];
#pragma unroll
    for (int nt = 0; nt < 2; ++nt) {
        const int col = waveN * 32 + nt * 16 + l16;  // local co 0..127
        bpp[nt] = sB + (col * 9) * 32 + half * 8;
    }

    const float* wbase = convw + (size_t)net * C_ * C_ * 9;

    for (int cb0 = 0; cb0 < C_; cb0 += 32) {     // 32-channel K chunk
        // ---- stage A: padded 10x10 patch, fused BN+ReLU, stored [b][yy][xx][ci]
        for (int r = wave; r < 20; r += 8) {     // (b_local, yy) rows
            const int bl = r / 10, yy = r % 10;
            const int ci = lane;
            const float sc = s_scale[cb0 + ci];
            const float sh = s_shift[cb0 + ci];
            const bool  yok = (unsigned)(yy - 1) < 8u;
            const float* xrow =
                x + (((size_t)(b0 + bl) * C_ + cb0 + ci) * H_ + (gy0 + yy - 1)) * W_
                  + gx0 - 1;
            unsigned short* arow = sA + ((bl * 10 + yy) * 10) * 32 + ci;
#pragma unroll
            for (int xx = 0; xx < 10; ++xx) {
                float val = 0.f;
                if (yok && (unsigned)(xx - 1) < 8u) {
                    const float h = xrow[xx] * sc + sh;
                    val = h > 0.f ? h : 0.f;
                }
                arow[xx * 32] = f2bf(val);
            }
        }

        // ---- stage B: 9 contiguous taps per (co,ci) -> sB[co][tap][ci]
        for (int it = 0; it < 16; ++it) {
            const int e  = it * 256 + tid;       // 0..4095
            const int co = e >> 5;               // local co 0..127
            const int ci = e & 31;
            const float* wp = wbase + ((size_t)(co0 + co) * C_ + cb0 + ci) * 9;
            unsigned short* bo = sB + (co * 9) * 32 + ci;
#pragma unroll
            for (int t = 0; t < 9; ++t)
                bo[t * 32] = f2bf(wp[t]);
            if (cb0 + 32 < C_)                   // next chunk (global_prefetch_b8)
                __builtin_prefetch(wp + 32 * 9, 0, 1);
        }
        __syncthreads();

        // ---- compute: 9 taps x 4M x 2N = 72 WMMAs per wave, no barriers
#pragma unroll
        for (int kq = 0; kq < 9; ++kq) {
            const int tap = ((kq / 3) * 10 + (kq % 3)) * 32;   // elems (imm)
            Frag bfr[2];
#pragma unroll
            for (int nt = 0; nt < 2; ++nt) {
                bfr[nt].q[0] = *(const u32x4*)(bpp[nt] + kq * 32);
                bfr[nt].q[1] = *(const u32x4*)(bpp[nt] + kq * 32 + 16);
            }
#pragma unroll
            for (int mt = 0; mt < 4; ++mt) {
                Frag a;
                a.q[0] = *(const u32x4*)(app[mt] + tap);
                a.q[1] = *(const u32x4*)(app[mt] + tap + 16);
#pragma unroll
                for (int nt = 0; nt < 2; ++nt)
                    acc[mt][nt] = __builtin_amdgcn_wmma_f32_16x16x32_bf16(
                        false, a.v, false, bfr[nt].v,
                        (short)0, acc[mt][nt], false, false);
            }
        }
        __syncthreads();
    }

    // ---- epilogue: + conv bias + residual, fp32 store ------------------
#pragma unroll
    for (int mt = 0; mt < 4; ++mt) {
#pragma unroll
        for (int nt = 0; nt < 2; ++nt) {
            const int co  = co0 + waveN * 32 + nt * 16 + l16;
            const float cbv = s_cb[co];
#pragma unroll
            for (int r = 0; r < 8; ++r) {
                const int m  = wgM * 128 + waveM * 64 + mt * 16 + half * 8 + r;
                const int b  = m >> 6;
                const int y  = (m >> 3) & 7;
                const int xx = m & 7;
                const size_t addr =
                    (((size_t)b * C_ + co) * H_ + (gy0 + y)) * W_ + (gx0 + xx);
                out[addr] = acc[mt][nt][r] + cbv + x[addr];
            }
        }
    }
}

extern "C" void kernel_launch(void* const* d_in, const int* in_sizes, int n_in,
                              void* d_out, int out_size, void* d_ws, size_t ws_size,
                              hipStream_t stream) {
    (void)in_sizes; (void)n_in; (void)d_ws; (void)ws_size; (void)out_size;
    const float* x     = (const float*)d_in[0];
    const float* gamma = (const float*)d_in[1];
    const float* beta  = (const float*)d_in[2];
    const float* mean  = (const float*)d_in[3];
    const float* var   = (const float*)d_in[4];
    const float* convw = (const float*)d_in[5];
    const float* convb = (const float*)d_in[6];
    float* out = (float*)d_out;

    // 64 regions x 8 M-blocks x 2 N-halves of the per-region implicit GEMM
    RegionLayer_33062658245169_kernel<<<dim3(64 * 8 * 2), dim3(256), 0, stream>>>(
        x, gamma, beta, mean, var, convw, convb, out);
}
